// GPS_58669253263737
// MI455X (gfx1250) — compile-verified, matrix operand order checked by reading, not run on hardware
//
#include <hip/hip_runtime.h>
#include <hip/hip_bf16.h>
#include <math.h>
#include <stdint.h>

// ---------------- problem constants (from reference) ----------------
constexpr int NN   = 8192;    // nodes
constexpr int BG   = 8;       // graphs
constexpr int NPG  = 1024;    // nodes per graph
constexpr int DD   = 256;     // hidden
constexpr int HH   = 8;       // heads
constexpr int DH   = 32;      // head dim
constexpr int LL   = 2;       // layers
constexpr int INC  = 128;
constexpr int OUTD = 64;
constexpr int EE   = 262144;
constexpr int DFF  = 512;
constexpr float EPS = 1e-5f;

typedef __attribute__((ext_vector_type(16))) __bf16 bf16x16;
typedef __attribute__((ext_vector_type(8)))  __bf16 bf16x8;
typedef __attribute__((ext_vector_type(8)))  float  f32x8;

static __device__ inline f32x8 wmma_bf16(bf16x16 a, bf16x16 b, f32x8 c) {
    return __builtin_amdgcn_wmma_f32_16x16x32_bf16(false, a, false, b, (short)0, c, false, false);
}

// Load 16 bf16 in A/B-operand per-lane layout: elements [0..7] at p, [8..15] at p+16.
static __device__ inline bf16x16 load_ab16(const __bf16* p) {
    bf16x8 lo = *(const bf16x8*)p;
    bf16x8 hi = *(const bf16x8*)(p + 16);
    bf16x16 r;
#pragma unroll
    for (int i = 0; i < 8; ++i) { r[i] = lo[i]; r[8 + i] = hi[i]; }
    return r;
}

// ---------------- elementwise helpers ----------------
__global__ void k_f32_to_bf16(const float* __restrict__ s, __bf16* __restrict__ d, int n) {
    int i = blockIdx.x * 256 + threadIdx.x;
    if (i < n) d[i] = (__bf16)s[i];
}

__global__ void k_zero_f32(float* __restrict__ p, int n) {
    int i = blockIdx.x * 256 + threadIdx.x;
    if (i < n) p[i] = 0.0f;
}

// ---------------- graph aggregation ----------------
__global__ void k_deg(const int* __restrict__ dst, float* __restrict__ deg, int e) {
    int i = blockIdx.x * 256 + threadIdx.x;
    if (i < e) atomicAdd(&deg[dst[i]], 1.0f);
}

__global__ void k_invdeg(const float* __restrict__ deg, float* __restrict__ inv, int n) {
    int i = blockIdx.x * 256 + threadIdx.x;
    if (i < n) inv[i] = 1.0f / fmaxf(deg[i], 1.0f);
}

__global__ void k_scatter(const int* __restrict__ src, const int* __restrict__ dst,
                          const float* __restrict__ h, float* __restrict__ agg, int e) {
    long long idx = (long long)blockIdx.x * 256 + threadIdx.x;
    if (idx >= (long long)e * 64) return;
    int ed = (int)(idx >> 6);
    int c  = (int)(idx & 63) << 2;
    int s = src[ed], d = dst[ed];
    const float* hp = h + (size_t)s * DD + c;
    float* ap = agg + (size_t)d * DD + c;
    atomicAdd(ap + 0, hp[0]);
    atomicAdd(ap + 1, hp[1]);
    atomicAdd(ap + 2, hp[2]);
    atomicAdd(ap + 3, hp[3]);
}

__global__ void k_scale_rows_bf(float* __restrict__ x, const float* __restrict__ invdeg,
                                __bf16* __restrict__ xbf, int rows, int cols) {
    long long i = (long long)blockIdx.x * 256 + threadIdx.x;
    if (i >= (long long)rows * cols) return;
    int r = (int)(i / cols);
    float v = x[i] * invdeg[r];
    x[i] = v;
    xbf[i] = (__bf16)v;
}

// ---------------- WMMA GEMM: C = A @ W^T (+bias)(+add0)(+add1), opt relu, opt bf16 dup ----------------
#define GF_RELU 1

__global__ __launch_bounds__(256)
void k_gemm_bf16(const __bf16* __restrict__ A, const __bf16* __restrict__ W,
                 const float* __restrict__ bias, const float* __restrict__ add0,
                 const float* __restrict__ add1, float* __restrict__ C,
                 __bf16* __restrict__ Cbf, int M, int N, int K, int flags) {
    const int lane = threadIdx.x & 31, widx = threadIdx.x >> 5;
    const int r = lane & 15, half = lane >> 4;
    const int ntn = N >> 6;                       // 64-wide N tiles per row
    const int gw = blockIdx.x * 8 + widx;
    const int m0 = (gw / ntn) << 4;
    const int n0 = (gw % ntn) << 6;
    if (m0 >= M) return;

    f32x8 acc0 = {}, acc1 = {}, acc2 = {}, acc3 = {};
    const __bf16* arow = A + (size_t)(m0 + r) * K + half * 8;
    const __bf16* w0 = W + (size_t)(n0 +  0 + r) * K + half * 8;
    const __bf16* w1 = W + (size_t)(n0 + 16 + r) * K + half * 8;
    const __bf16* w2 = W + (size_t)(n0 + 32 + r) * K + half * 8;
    const __bf16* w3 = W + (size_t)(n0 + 48 + r) * K + half * 8;

    for (int k0 = 0; k0 < K; k0 += 32) {
        bf16x16 a  = load_ab16(arow + k0);
        bf16x16 b0 = load_ab16(w0 + k0);
        bf16x16 b1 = load_ab16(w1 + k0);
        bf16x16 b2 = load_ab16(w2 + k0);
        bf16x16 b3 = load_ab16(w3 + k0);
        acc0 = wmma_bf16(a, b0, acc0);
        acc1 = wmma_bf16(a, b1, acc1);
        acc2 = wmma_bf16(a, b2, acc2);
        acc3 = wmma_bf16(a, b3, acc3);
    }

    f32x8 accs[4] = {acc0, acc1, acc2, acc3};
#pragma unroll
    for (int j = 0; j < 4; ++j) {
        const int n = n0 + j * 16 + r;
        const float bv = bias ? bias[n] : 0.0f;
#pragma unroll
        for (int v = 0; v < 8; ++v) {
            const int m = m0 + half * 8 + v;
            const size_t idx = (size_t)m * N + n;
            float val = accs[j][v] + bv;
            if (add0) val += add0[idx];
            if (add1) val += add1[idx];
            if (flags & GF_RELU) val = fmaxf(val, 0.0f);
            C[idx] = val;
            if (Cbf) Cbf[idx] = (__bf16)val;
        }
    }
}

// ---------------- BatchNorm (training-mode batch stats over nodes) ----------------
__global__ __launch_bounds__(256)
void k_bn_stats(const float* __restrict__ x, float* __restrict__ mean,
                float* __restrict__ var, int rows, int cols) {
    __shared__ float s1[256], s2[256];
    const int d = blockIdx.x, t = threadIdx.x;
    float a = 0.0f, b = 0.0f;
    for (int i = t; i < rows; i += 256) {
        float v = x[(size_t)i * cols + d];
        a += v; b += v * v;
    }
    s1[t] = a; s2[t] = b;
    __syncthreads();
    for (int off = 128; off > 0; off >>= 1) {
        if (t < off) { s1[t] += s1[t + off]; s2[t] += s2[t + off]; }
        __syncthreads();
    }
    if (t == 0) {
        float m = s1[0] / rows;
        mean[d] = m;
        var[d] = s2[0] / rows - m * m;
    }
}

#define BNF_RELU_RESID 1   // y = addsrc + relu(bn(x)); else y = bn(x) + (addsrc?addsrc:0)

__global__ void k_bn_apply(const float* __restrict__ x, const float* __restrict__ mean,
                           const float* __restrict__ var, const float* __restrict__ w,
                           const float* __restrict__ b, const float* __restrict__ addsrc,
                           float* __restrict__ y, __bf16* __restrict__ ybf,
                           int rows, int cols, int flags) {
    long long i = (long long)blockIdx.x * 256 + threadIdx.x;
    if (i >= (long long)rows * cols) return;
    int d = (int)(i % cols);
    float xn = (x[i] - mean[d]) * rsqrtf(var[d] + EPS) * w[d] + b[d];
    float val;
    if (flags & BNF_RELU_RESID) val = addsrc[i] + fmaxf(xn, 0.0f);
    else                        val = xn + (addsrc ? addsrc[i] : 0.0f);
    y[i] = val;
    if (ybf) ybf[i] = (__bf16)val;
}

// ---------------- flash attention: wave = 16 queries x 1 head, K-depth = DH = 32 ----------------
// V tiles are staged global->LDS with CDNA5 async-DMA (global_load_async_to_lds_b128,
// ASYNCcnt), issued before the S/softmax work of the same chunk so the copy overlaps
// compute; reads then come from LDS instead of 3KB-strided uncoalesced global loads.
constexpr int VPITCH = 36;  // f32 pitch per key row: 144B, 16B-aligned, bank-spread

__global__ __launch_bounds__(256)
void k_attn(const float* __restrict__ qkv, float* __restrict__ o, __bf16* __restrict__ obf) {
    __shared__ __bf16 smemP[8][16 * 32];                     // per-wave P tile (16 q x 32 k)
    __shared__ __attribute__((aligned(16))) float smemV[8][32 * VPITCH]; // per-wave V stage
    const int lane = threadIdx.x & 31, widx = threadIdx.x >> 5;
    const int r = lane & 15, half = lane >> 4;
    const int gw = blockIdx.x * 8 + widx;      // 4096 waves total
    const int head = gw & 7;
    const int qt = gw >> 3;                    // 0..511 query tiles
    const int g = qt >> 6;                     // graph id
    const int qbase = qt << 4;                 // first query node (graphs contiguous)
    const int kstart = g << 10;
    const float scale = 0.17677669529663687f;  // 1/sqrt(32)

    // Q tile as A operand (pre-scaled)
    const float* qp = qkv + (size_t)(qbase + r) * 768 + head * 32 + half * 8;
    bf16x16 aq;
#pragma unroll
    for (int i = 0; i < 8; ++i) {
        aq[i]     = (__bf16)(qp[i] * scale);
        aq[8 + i] = (__bf16)(qp[16 + i] * scale);
    }

    f32x8 accO0 = {}, accO1 = {};
    float rmax[8], rsum[8];
#pragma unroll
    for (int v = 0; v < 8; ++v) { rmax[v] = -INFINITY; rsum[v] = 0.0f; }

    __bf16* P = &smemP[widx][0];
    float*  Vl = &smemV[widx][0];
    const uint32_t vlds = (uint32_t)(uintptr_t)Vl + (uint32_t)(lane * (VPITCH * 4));

    for (int kc = 0; kc < NPG; kc += 32) {
        // ---- async-stage V chunk (32 keys x 32 dh f32): lane owns key==lane, 8x b128 ----
        {
            // WAR guard: previous chunk's LDS reads must have landed in VGPRs.
            asm volatile("s_wait_dscnt 0x0" ::: "memory");
            const float* grow = qkv + (size_t)(kstart + kc + lane) * 768 + 512 + head * 32;
#pragma unroll
            for (int j = 0; j < 8; ++j) {
                asm volatile("global_load_async_to_lds_b128 %0, %1, off"
                             :: "v"(vlds + (uint32_t)(j * 16)), "v"(grow + j * 4)
                             : "memory");
            }
        }

        // ---- S = (Q*scale) K^T for 32 keys: two 16x16x32 WMMAs (overlaps the V DMA) ----
        const float* kp0 = qkv + (size_t)(kstart + kc +      r) * 768 + 256 + head * 32 + half * 8;
        const float* kp1 = qkv + (size_t)(kstart + kc + 16 + r) * 768 + 256 + head * 32 + half * 8;
        bf16x16 bk0, bk1;
#pragma unroll
        for (int i = 0; i < 8; ++i) {
            bk0[i] = (__bf16)kp0[i];  bk0[8 + i] = (__bf16)kp0[16 + i];
            bk1[i] = (__bf16)kp1[i];  bk1[8 + i] = (__bf16)kp1[16 + i];
        }
        f32x8 z = {};
        f32x8 s0 = wmma_bf16(aq, bk0, z);
        f32x8 s1 = wmma_bf16(aq, bk1, z);

        // ---- online softmax; row quantities reduce across the 16-lane half ----
#pragma unroll
        for (int v = 0; v < 8; ++v) {
            float mx = fmaxf(s0[v], s1[v]);
            mx = fmaxf(mx, __shfl_xor(mx, 1, 32));
            mx = fmaxf(mx, __shfl_xor(mx, 2, 32));
            mx = fmaxf(mx, __shfl_xor(mx, 4, 32));
            mx = fmaxf(mx, __shfl_xor(mx, 8, 32));
            float nm = fmaxf(rmax[v], mx);
            float corr = __expf(rmax[v] - nm);
            float p0 = __expf(s0[v] - nm);
            float p1 = __expf(s1[v] - nm);
            float ps = p0 + p1;
            ps += __shfl_xor(ps, 1, 32);
            ps += __shfl_xor(ps, 2, 32);
            ps += __shfl_xor(ps, 4, 32);
            ps += __shfl_xor(ps, 8, 32);
            rsum[v] = rsum[v] * corr + ps;
            rmax[v] = nm;
            accO0[v] *= corr;
            accO1[v] *= corr;
            const int row = half * 8 + v;       // query row 0..15
            P[row * 32 + r]      = (__bf16)p0;  // key col r
            P[row * 32 + 16 + r] = (__bf16)p1;  // key col 16+r
        }

        // ---- re-read P in A-operand layout (per-wave LDS, in-order DS) ----
        bf16x16 ap;
#pragma unroll
        for (int i = 0; i < 8; ++i) {
            ap[i]     = P[r * 32 + half * 8 + i];
            ap[8 + i] = P[r * 32 + 16 + half * 8 + i];
        }

        // ---- wait for the V DMA, then build B operands from LDS ----
        asm volatile("s_wait_asynccnt 0x0" ::: "memory");
        bf16x16 bv0, bv1;
#pragma unroll
        for (int i = 0; i < 8; ++i) {
            const int kA = half * 8 + i, kB = 16 + half * 8 + i;
            bv0[i]     = (__bf16)Vl[kA * VPITCH + r];
            bv0[8 + i] = (__bf16)Vl[kB * VPITCH + r];
            bv1[i]     = (__bf16)Vl[kA * VPITCH + 16 + r];
            bv1[8 + i] = (__bf16)Vl[kB * VPITCH + 16 + r];
        }
        accO0 = wmma_bf16(ap, bv0, accO0);
        accO1 = wmma_bf16(ap, bv1, accO1);
    }

    // ---- epilogue: O /= rowsum ----
#pragma unroll
    for (int v = 0; v < 8; ++v) {
        const float inv = 1.0f / rsum[v];
        const int node = qbase + half * 8 + v;
        const size_t base = (size_t)node * DD + head * 32;
        float o0 = accO0[v] * inv, o1 = accO1[v] * inv;
        o[base + r] = o0;        obf[base + r] = (__bf16)o0;
        o[base + 16 + r] = o1;   obf[base + 16 + r] = (__bf16)o1;
    }
}

// ---------------- global mean pool + output projection ----------------
__global__ __launch_bounds__(256)
void k_pool_out(const float* __restrict__ h, const float* __restrict__ w_out,
                const float* __restrict__ b_out, float* __restrict__ out) {
    __shared__ float pooled[DD];
    const int g = blockIdx.x, t = threadIdx.x;
    const float* base = h + (size_t)g * NPG * DD;
    float s = 0.0f;
    for (int i = 0; i < NPG; ++i) s += base[(size_t)i * DD + t];
    pooled[t] = s * (1.0f / NPG);
    __syncthreads();
    if (t < OUTD) {
        float acc = b_out[t];
        for (int d = 0; d < DD; ++d) acc += pooled[d] * w_out[(size_t)t * DD + d];
        out[g * OUTD + t] = acc;
    }
}

// ---------------- host-side orchestration ----------------
static inline void gemm(const __bf16* A, const __bf16* W, const float* bias,
                        const float* add0, const float* add1, float* C, __bf16* Cbf,
                        int M, int N, int K, int flags, hipStream_t s) {
    int waves = (M / 16) * (N / 64);
    int blocks = (waves + 7) / 8;
    k_gemm_bf16<<<blocks, 256, 0, s>>>(A, W, bias, add0, add1, C, Cbf, M, N, K, flags);
}

static inline int cdiv(long long a, int b) { return (int)((a + b - 1) / b); }

extern "C" void kernel_launch(void* const* d_in, const int* in_sizes, int n_in,
                              void* d_out, int out_size, void* d_ws, size_t ws_size,
                              hipStream_t stream) {
    (void)in_sizes; (void)n_in; (void)out_size; (void)ws_size;

    const float* x       = (const float*)d_in[0];
    const int*   ei      = (const int*)d_in[1];
    const float* w_in    = (const float*)d_in[3];
    const float* b_in    = (const float*)d_in[4];
    const float* sage_wl = (const float*)d_in[5];
    const float* sage_bl = (const float*)d_in[6];
    const float* sage_wr = (const float*)d_in[7];
    const float* attn_iw = (const float*)d_in[8];
    const float* attn_ib = (const float*)d_in[9];
    const float* attn_ow = (const float*)d_in[10];
    const float* attn_ob = (const float*)d_in[11];
    const float* n1_w = (const float*)d_in[12];
    const float* n1_b = (const float*)d_in[13];
    const float* n2_w = (const float*)d_in[14];
    const float* n2_b = (const float*)d_in[15];
    const float* n3_w = (const float*)d_in[16];
    const float* n3_b = (const float*)d_in[17];
    const float* mlp_w1 = (const float*)d_in[18];
    const float* mlp_b1 = (const float*)d_in[19];
    const float* mlp_w2 = (const float*)d_in[20];
    const float* mlp_b2 = (const float*)d_in[21];
    const float* bn_w = (const float*)d_in[22];
    const float* bn_b = (const float*)d_in[23];
    const float* w_out = (const float*)d_in[24];
    const float* b_out = (const float*)d_in[25];

    const int* esrc = ei;
    const int* edst = ei + EE;

    // ---- workspace carve-out ----
    char* p = (char*)d_ws;
    auto alloc = [&](size_t bytes) -> char* {
        char* r = p; p += (bytes + 255) & ~(size_t)255; return r;
    };
    float*  h       = (float*)alloc((size_t)NN * DD * 4);
    __bf16* h_bf    = (__bf16*)alloc((size_t)NN * DD * 2);
    float*  bufA    = (float*)alloc((size_t)NN * DD * 4);   // loc_pre / out0
    __bf16* bufA_bf = (__bf16*)alloc((size_t)NN * DD * 2);  // agg_bf / out0_bf
    float*  bufB    = (float*)alloc((size_t)NN * DD * 4);   // tmp / o2 / out1
    float*  bufC    = (float*)alloc((size_t)NN * DD * 4);   // agg / loc / out2
    char*   qkv_rg  = alloc((size_t)NN * 3 * DD * 4);       // qkv; later hid + hid_bf
    float*  qkv     = (float*)qkv_rg;
    float*  hid     = (float*)qkv_rg;                       // reuse after attention
    __bf16* hid_bf  = (__bf16*)(qkv_rg + (size_t)NN * DFF * 4);
    float*  o_buf   = (float*)alloc((size_t)NN * DD * 4);
    __bf16* o_bf    = (__bf16*)alloc((size_t)NN * DD * 2);
    float*  deg     = (float*)alloc((size_t)NN * 4);
    float*  invdeg  = (float*)alloc((size_t)NN * 4);
    float*  mean    = (float*)alloc(DD * 4);
    float*  var     = (float*)alloc(DD * 4);
    __bf16* x_bf       = (__bf16*)alloc((size_t)NN * INC * 2);
    __bf16* w_in_bf    = (__bf16*)alloc((size_t)DD * INC * 2);
    __bf16* sage_wl_bf = (__bf16*)alloc((size_t)LL * DD * DD * 2);
    __bf16* sage_wr_bf = (__bf16*)alloc((size_t)LL * DD * DD * 2);
    __bf16* attn_iw_bf = (__bf16*)alloc((size_t)LL * 3 * DD * DD * 2);
    __bf16* attn_ow_bf = (__bf16*)alloc((size_t)LL * DD * DD * 2);
    __bf16* mlp_w1_bf  = (__bf16*)alloc((size_t)LL * DFF * DD * 2);
    __bf16* mlp_w2_bf  = (__bf16*)alloc((size_t)LL * DD * DFF * 2);

    // ---- weight / input bf16 conversion ----
    auto cvt = [&](const float* s, __bf16* d, long long n) {
        k_f32_to_bf16<<<cdiv(n, 256), 256, 0, stream>>>(s, d, (int)n);
    };
    cvt(x, x_bf, (long long)NN * INC);
    cvt(w_in, w_in_bf, (long long)DD * INC);
    cvt(sage_wl, sage_wl_bf, (long long)LL * DD * DD);
    cvt(sage_wr, sage_wr_bf, (long long)LL * DD * DD);
    cvt(attn_iw, attn_iw_bf, (long long)LL * 3 * DD * DD);
    cvt(attn_ow, attn_ow_bf, (long long)LL * DD * DD);
    cvt(mlp_w1, mlp_w1_bf, (long long)LL * DFF * DD);
    cvt(mlp_w2, mlp_w2_bf, (long long)LL * DD * DFF);

    // ---- degrees ----
    k_zero_f32<<<cdiv(NN, 256), 256, 0, stream>>>(deg, NN);
    k_deg<<<cdiv(EE, 256), 256, 0, stream>>>(edst, deg, EE);
    k_invdeg<<<cdiv(NN, 256), 256, 0, stream>>>(deg, invdeg, NN);

    // ---- input projection: h = x @ w_in^T + b_in ----
    gemm(x_bf, w_in_bf, b_in, nullptr, nullptr, h, h_bf, NN, DD, INC, 0, stream);

    const long long ND = (long long)NN * DD;
    const int ndBlocks = cdiv(ND, 256);

    for (int l = 0; l < LL; ++l) {
        // ---- SAGE mean aggregation: agg (bufC) ----
        k_zero_f32<<<ndBlocks, 256, 0, stream>>>(bufC, (int)ND);
        k_scatter<<<cdiv((long long)EE * 64, 256), 256, 0, stream>>>(esrc, edst, h, bufC, EE);
        k_scale_rows_bf<<<ndBlocks, 256, 0, stream>>>(bufC, invdeg, bufA_bf, NN, DD);

        // tmp = agg @ wl^T + bl  (bufB)
        gemm(bufA_bf, sage_wl_bf + (size_t)l * DD * DD, sage_bl + (size_t)l * DD,
             nullptr, nullptr, bufB, nullptr, NN, DD, DD, 0, stream);
        // loc_pre = h @ wr^T + tmp + h  (bufA)
        gemm(h_bf, sage_wr_bf + (size_t)l * DD * DD, nullptr,
             bufB, h, bufA, nullptr, NN, DD, DD, 0, stream);
        // loc = BN1(loc_pre)  (bufC)
        k_bn_stats<<<DD, 256, 0, stream>>>(bufA, mean, var, NN, DD);
        k_bn_apply<<<ndBlocks, 256, 0, stream>>>(bufA, mean, var,
            n1_w + (size_t)l * DD, n1_b + (size_t)l * DD, nullptr, bufC, nullptr, NN, DD, 0);

        // ---- attention branch ----
        gemm(h_bf, attn_iw_bf + (size_t)l * 3 * DD * DD, attn_ib + (size_t)l * 3 * DD,
             nullptr, nullptr, qkv, nullptr, NN, 3 * DD, DD, 0, stream);
        k_attn<<<512, 256, 0, stream>>>(qkv, o_buf, o_bf);
        // o2 = o @ ow^T + ob + h  (bufB)
        gemm(o_bf, attn_ow_bf + (size_t)l * DD * DD, attn_ob + (size_t)l * DD,
             h, nullptr, bufB, nullptr, NN, DD, DD, 0, stream);
        // glob = BN2(o2); out0 = loc + glob  (bufA + bufA_bf)
        k_bn_stats<<<DD, 256, 0, stream>>>(bufB, mean, var, NN, DD);
        k_bn_apply<<<ndBlocks, 256, 0, stream>>>(bufB, mean, var,
            n2_w + (size_t)l * DD, n2_b + (size_t)l * DD, bufC, bufA, bufA_bf, NN, DD, 0);

        // ---- MLP: hid = relu(out0 @ w1^T + b1); out1 = hid @ w2^T + b2 + out0 ----
        gemm(bufA_bf, mlp_w1_bf + (size_t)l * DFF * DD, mlp_b1 + (size_t)l * DFF,
             nullptr, nullptr, hid, hid_bf, NN, DFF, DD, GF_RELU, stream);
        gemm(hid_bf, mlp_w2_bf + (size_t)l * DD * DFF, mlp_b2 + (size_t)l * DD,
             bufA, nullptr, bufB, nullptr, NN, DD, DFF, 0, stream);

        // out2 = BN3(out1)  (bufC)
        k_bn_stats<<<DD, 256, 0, stream>>>(bufB, mean, var, NN, DD);
        k_bn_apply<<<ndBlocks, 256, 0, stream>>>(bufB, mean, var,
            n3_w + (size_t)l * DD, n3_b + (size_t)l * DD, nullptr, bufC, nullptr, NN, DD, 0);

        // h = h + relu(BN_outer(out2))  (+ bf16 dup)
        k_bn_stats<<<DD, 256, 0, stream>>>(bufC, mean, var, NN, DD);
        k_bn_apply<<<ndBlocks, 256, 0, stream>>>(bufC, mean, var,
            bn_w + (size_t)l * DD, bn_b + (size_t)l * DD, h, h, h_bf, NN, DD, BNF_RELU_RESID);
    }

    // ---- pool + output projection ----
    k_pool_out<<<BG, 256, 0, stream>>>(h, w_out, b_out, (float*)d_out);
}
